// SAGE_76347338654181
// MI455X (gfx1250) — compile-verified, matrix operand order checked by reading, not run on hardware
//
#include <hip/hip_runtime.h>
#include <hip/hip_bf16.h>
#include <math.h>

// ---------------------------------------------------------------------------
// GraphSAGE 3-layer inference for MI455X (gfx1250, wave32, WMMA).
//   layer l: out = relu( [mean_agg | x_tgt] @ [Wl; Wr] + b )   (fused K-concat GEMM)
// A and W are pre-swizzled into WMMA fragment order so the GEMM inner loop is
// pure b128 loads + v_wmma_f32_16x16x32_bf16. Aggregation uses HW fp32 global
// atomics; final log_softmax is a wave32 shuffle reduction.
// ---------------------------------------------------------------------------

typedef __attribute__((ext_vector_type(16))) __bf16 v16bf;
typedef __attribute__((ext_vector_type(8)))  float  v8f;

// Problem constants (from the reference).
constexpr int cN0 = 600000, cN1 = 100000, cN2 = 20000, cN3 = 4096;
constexpr int cE0 = 1500000, cE1 = 200000, cE2 = 40960;
constexpr int cDIN = 100, cDH = 256, cDOUT = 47;

// Padded tile dims (M multiple of 128, N multiple of 64, K multiple of 32).
constexpr int MP0 = 100096;   // ceil(100000/128)*128
constexpr int MP1 = 20096;    // ceil(20000/128)*128
constexpr int MP2 = 4096;
constexpr int KA0 = 256;      // [mean(100->128) | x(100->128)]
constexpr int KA12 = 512;     // [mean(256) | h(256)]
constexpr int NP01 = 256;
constexpr int NP2 = 64;       // 47 -> 64

// ---------------------------------------------------------------------------
// WMMA 16-bit fragment swizzle (ISA 7.12.2):
//   within a 32-wide K chunk: lanehalf = (kk>>3)&1, pos = ((kk&16)>>1)|(kk&7)
//   A tile 16x32: lane = lanehalf*16 + row ; B tile 32x16: lane = lanehalf*16 + col
// Packed storage: A = [mt][kt][lane][pos], W = [kt][nt][lane][pos] (16 bf16/lane).
// ---------------------------------------------------------------------------
__device__ __forceinline__ size_t a_frag_idx(int i, int k, int K) {
    int mt = i >> 4, row = i & 15;
    int kt = k >> 5, kk = k & 31;
    int lanehalf = (kk >> 3) & 1;
    int pos = ((kk & 16) >> 1) | (kk & 7);
    return ((size_t)(mt * (K >> 5) + kt) * 32 + lanehalf * 16 + row) * 16 + pos;
}
__device__ __forceinline__ size_t b_frag_idx(int k, int n, int Npad) {
    int nt = n >> 4, col = n & 15;
    int kt = k >> 5, kk = k & 31;
    int lanehalf = (kk >> 3) & 1;
    int pos = ((kk & 16) >> 1) | (kk & 7);
    return ((size_t)(kt * (Npad >> 4) + nt) * 32 + lanehalf * 16 + col) * 16 + pos;
}

// ---------------------------------------------------------------------------
// Aggregation: per-edge scatter-add with HW fp32 atomics (L2-resident target).
// ---------------------------------------------------------------------------
__global__ void sage_scatter_kernel(const float* __restrict__ x,
                                    const int* __restrict__ src,
                                    const int* __restrict__ dst,
                                    float* __restrict__ agg, int D) {
    int e = blockIdx.x;
    int j = threadIdx.x;
    int s = src[e];
    int d = dst[e];
    if (j < D) {
        unsafeAtomicAdd(&agg[(size_t)d * D + j], x[(size_t)s * D + j]);
    }
}

__global__ void sage_count_kernel(const int* __restrict__ dst,
                                  float* __restrict__ cnt, int E) {
    int e = blockIdx.x * blockDim.x + threadIdx.x;
    if (e < E) unsafeAtomicAdd(&cnt[dst[e]], 1.0f);
}

// ---------------------------------------------------------------------------
// Pack A = [mean | x_tgt] in bf16, directly in fragment order. Pad zones
// are covered by the zero memset (swizzle is a bijection over the buffer).
// ---------------------------------------------------------------------------
__global__ void sage_pack_a0_kernel(const float* __restrict__ agg,
                                    const float* __restrict__ cnt,
                                    const float* __restrict__ x,
                                    __bf16* __restrict__ A) {
    int i = blockIdx.x;            // node < N1
    int j = threadIdx.x;           // 0..127
    if (j >= cDIN) return;
    float rcp = 1.0f / fmaxf(cnt[i], 1.0f);
    A[a_frag_idx(i, j, KA0)]       = (__bf16)(agg[(size_t)i * cDIN + j] * rcp);
    A[a_frag_idx(i, 128 + j, KA0)] = (__bf16)x[(size_t)i * cDIN + j];
}

__global__ void sage_pack_a_dh_kernel(const float* __restrict__ agg,
                                      const float* __restrict__ cnt,
                                      const float* __restrict__ h,
                                      __bf16* __restrict__ A) {
    int i = blockIdx.x;            // target node
    int j = threadIdx.x;           // 0..255
    float rcp = 1.0f / fmaxf(cnt[i], 1.0f);
    A[a_frag_idx(i, j, KA12)]       = (__bf16)(agg[(size_t)i * cDH + j] * rcp);
    A[a_frag_idx(i, cDH + j, KA12)] = (__bf16)h[(size_t)i * cDH + j];
}

// Wcat = [Wl ; 0-pad ; Wr ; 0-pad] in fragment order (full coverage, no memset
// needed). Also pads bias.
__global__ void sage_pack_w_kernel(const float* __restrict__ Wl,
                                   const float* __restrict__ Wr,
                                   const float* __restrict__ b,
                                   __bf16* __restrict__ Wpk,
                                   float* __restrict__ biasPad,
                                   int Kin, int Khalf, int N, int Npad) {
    int k = blockIdx.x;            // 0 .. 2*Khalf-1
    int n = threadIdx.x;           // 0 .. Npad-1
    float v = 0.0f;
    if (n < N) {
        if (k < Kin)                            v = Wl[(size_t)k * N + n];
        else if (k >= Khalf && k < Khalf + Kin) v = Wr[(size_t)(k - Khalf) * N + n];
    }
    Wpk[b_frag_idx(k, n, Npad)] = (__bf16)v;
    if (k == 0) biasPad[n] = (n < N) ? b[n] : 0.0f;
}

// ---------------------------------------------------------------------------
// WMMA GEMM on pre-swizzled fragments:
//   C[Mpad x Npad] = A[Mpad x K] @ W[K x Npad] + bias (+ReLU)
// Block = 256 threads = 8 waves; each wave owns a 16x64 strip (4 accumulators).
// Inner loop: 1 A-fragment + 4 B-fragments (32B vector loads) + 4 WMMAs.
// ---------------------------------------------------------------------------
__global__ void sage_gemm_bf16_wmma(const __bf16* __restrict__ Apk,
                                    const __bf16* __restrict__ Wpk,
                                    const float* __restrict__ bias,
                                    float* __restrict__ C,
                                    int K, int Npad, int relu) {
    const int wave = threadIdx.x >> 5;
    const int lane = threadIdx.x & 31;
    const int mt     = blockIdx.x * 8 + wave;   // 16-row strip
    const int ntBase = blockIdx.y * 4;          // 4 x 16-col tiles
    const int KT = K >> 5;
    const int NT = Npad >> 4;

    v8f zero = {};
    v8f acc0 = zero, acc1 = zero, acc2 = zero, acc3 = zero;

    const __bf16* aPtr = Apk + ((size_t)mt * KT * 32 + lane) * 16;
    const __bf16* bPtr = Wpk + ((size_t)ntBase * 32 + lane) * 16;

    for (int kt = 0; kt < KT; ++kt) {
        __builtin_prefetch(aPtr + (size_t)(kt + 1) * 512, 0, 0);  // global_prefetch
        v16bf a = *(const v16bf*)(aPtr + (size_t)kt * 512);
        const __bf16* bk = bPtr + (size_t)kt * NT * 512;
        v16bf b0 = *(const v16bf*)(bk);
        v16bf b1 = *(const v16bf*)(bk + 512);
        v16bf b2 = *(const v16bf*)(bk + 1024);
        v16bf b3 = *(const v16bf*)(bk + 1536);
        acc0 = __builtin_amdgcn_wmma_f32_16x16x32_bf16(false, a, false, b0, (short)0, acc0, false, false);
        acc1 = __builtin_amdgcn_wmma_f32_16x16x32_bf16(false, a, false, b1, (short)0, acc1, false, false);
        acc2 = __builtin_amdgcn_wmma_f32_16x16x32_bf16(false, a, false, b2, (short)0, acc2, false, false);
        acc3 = __builtin_amdgcn_wmma_f32_16x16x32_bf16(false, a, false, b3, (short)0, acc3, false, false);
    }

    // C/D layout: VGPR r -> row (r + 8*hi), col = 16*tile + (lane&15).
    const int mrow = lane & 15;
    const int hi   = lane >> 4;
#pragma unroll
    for (int t = 0; t < 4; ++t) {
        v8f acc = (t == 0) ? acc0 : (t == 1) ? acc1 : (t == 2) ? acc2 : acc3;
        int col = (ntBase + t) * 16 + mrow;
        float bv = bias[col];
#pragma unroll
        for (int r = 0; r < 8; ++r) {
            int row = mt * 16 + r + 8 * hi;
            float v = acc[r] + bv;
            if (relu) v = fmaxf(v, 0.0f);
            C[(size_t)row * Npad + col] = v;
        }
    }
}

// ---------------------------------------------------------------------------
// log_softmax over 47 classes: one wave per row, wave32 shuffle reductions.
// ---------------------------------------------------------------------------
__global__ void sage_logsoftmax_kernel(const float* __restrict__ logits,
                                       float* __restrict__ out) {
    int row  = blockIdx.x;
    int lane = threadIdx.x;               // 0..31
    const float* lp = logits + (size_t)row * NP2;
    float ninf = -__builtin_inff();
    float v0 = (lane < cDOUT)      ? lp[lane]      : ninf;
    float v1 = (lane + 32 < cDOUT) ? lp[lane + 32] : ninf;
    float m = fmaxf(v0, v1);
#pragma unroll
    for (int off = 16; off > 0; off >>= 1) m = fmaxf(m, __shfl_xor(m, off, 32));
    float s = 0.0f;
    if (lane < cDOUT)      s += expf(v0 - m);
    if (lane + 32 < cDOUT) s += expf(v1 - m);
#pragma unroll
    for (int off = 16; off > 0; off >>= 1) s += __shfl_xor(s, off, 32);
    float lse = m + logf(s);
    if (lane < cDOUT)      out[(size_t)row * cDOUT + lane]      = v0 - lse;
    if (lane + 32 < cDOUT) out[(size_t)row * cDOUT + lane + 32] = v1 - lse;
}

// ---------------------------------------------------------------------------
extern "C" void kernel_launch(void* const* d_in, const int* in_sizes, int n_in,
                              void* d_out, int out_size, void* d_ws, size_t ws_size,
                              hipStream_t stream) {
    const float* x       = (const float*)d_in[0];
    const int*   ei0_src = (const int*)d_in[1];
    const int*   ei0_dst = (const int*)d_in[2];
    const int*   ei1_src = (const int*)d_in[3];
    const int*   ei1_dst = (const int*)d_in[4];
    const int*   ei2_src = (const int*)d_in[5];
    const int*   ei2_dst = (const int*)d_in[6];
    const float* Wl0 = (const float*)d_in[7];
    const float* Wr0 = (const float*)d_in[8];
    const float* b0  = (const float*)d_in[9];
    const float* Wl1 = (const float*)d_in[10];
    const float* Wr1 = (const float*)d_in[11];
    const float* b1  = (const float*)d_in[12];
    const float* Wl2 = (const float*)d_in[13];
    const float* Wr2 = (const float*)d_in[14];
    const float* b2  = (const float*)d_in[15];
    float* out = (float*)d_out;

    // ---- workspace layout (zero-prefix first, one memset covers it) ----
    char* ws = (char*)d_ws;
    size_t off = 0;
    auto alloc = [&](size_t bytes) -> void* {
        void* p = ws + off;
        off = (off + bytes + 255) & ~(size_t)255;
        return p;
    };
    float*  agg0 = (float*)alloc((size_t)cN1 * cDIN * 4);
    float*  cnt0 = (float*)alloc((size_t)cN1 * 4);
    float*  agg1 = (float*)alloc((size_t)cN2 * cDH * 4);
    float*  cnt1 = (float*)alloc((size_t)cN2 * 4);
    float*  agg2 = (float*)alloc((size_t)cN3 * cDH * 4);
    float*  cnt2 = (float*)alloc((size_t)cN3 * 4);
    __bf16* A0   = (__bf16*)alloc((size_t)MP0 * KA0 * 2);
    __bf16* A1   = (__bf16*)alloc((size_t)MP1 * KA12 * 2);
    __bf16* A2   = (__bf16*)alloc((size_t)MP2 * KA12 * 2);
    size_t zeroBytes = off;                 // everything above must start at 0
    float*  h1     = (float*)alloc((size_t)MP0 * NP01 * 4);
    float*  h2     = (float*)alloc((size_t)MP1 * NP01 * 4);
    float*  logits = (float*)alloc((size_t)MP2 * NP2 * 4);
    __bf16* Wc0 = (__bf16*)alloc((size_t)KA0 * NP01 * 2);
    __bf16* Wc1 = (__bf16*)alloc((size_t)KA12 * NP01 * 2);
    __bf16* Wc2 = (__bf16*)alloc((size_t)KA12 * NP2 * 2);
    float*  bp0 = (float*)alloc(NP01 * 4);
    float*  bp1 = (float*)alloc(NP01 * 4);
    float*  bp2 = (float*)alloc(NP2 * 4);
    (void)ws_size; (void)n_in; (void)in_sizes; (void)out_size;

    hipMemsetAsync(d_ws, 0, zeroBytes, stream);

    // ---- pack weights into fragment order (independent of activations) ----
    sage_pack_w_kernel<<<dim3(KA0),  dim3(NP01), 0, stream>>>(Wl0, Wr0, b0, Wc0, bp0, cDIN, KA0 / 2, cDH,  NP01);
    sage_pack_w_kernel<<<dim3(KA12), dim3(NP01), 0, stream>>>(Wl1, Wr1, b1, Wc1, bp1, cDH,  KA12 / 2, cDH,  NP01);
    sage_pack_w_kernel<<<dim3(KA12), dim3(NP2),  0, stream>>>(Wl2, Wr2, b2, Wc2, bp2, cDH,  KA12 / 2, cDOUT, NP2);

    // ---- layer 0 ----
    sage_scatter_kernel<<<dim3(cE0), dim3(128), 0, stream>>>(x, ei0_src, ei0_dst, agg0, cDIN);
    sage_count_kernel<<<dim3((cE0 + 255) / 256), dim3(256), 0, stream>>>(ei0_dst, cnt0, cE0);
    sage_pack_a0_kernel<<<dim3(cN1), dim3(128), 0, stream>>>(agg0, cnt0, x, A0);
    sage_gemm_bf16_wmma<<<dim3(MP0 / 128, NP01 / 64), dim3(256), 0, stream>>>(A0, Wc0, bp0, h1, KA0, NP01, 1);

    // ---- layer 1 ----
    sage_scatter_kernel<<<dim3(cE1), dim3(256), 0, stream>>>(h1, ei1_src, ei1_dst, agg1, cDH);
    sage_count_kernel<<<dim3((cE1 + 255) / 256), dim3(256), 0, stream>>>(ei1_dst, cnt1, cE1);
    sage_pack_a_dh_kernel<<<dim3(cN2), dim3(256), 0, stream>>>(agg1, cnt1, h1, A1);
    sage_gemm_bf16_wmma<<<dim3(MP1 / 128, NP01 / 64), dim3(256), 0, stream>>>(A1, Wc1, bp1, h2, KA12, NP01, 1);

    // ---- layer 2 ----
    sage_scatter_kernel<<<dim3(cE2), dim3(256), 0, stream>>>(h2, ei2_src, ei2_dst, agg2, cDH);
    sage_count_kernel<<<dim3((cE2 + 255) / 256), dim3(256), 0, stream>>>(ei2_dst, cnt2, cE2);
    sage_pack_a_dh_kernel<<<dim3(cN3), dim3(256), 0, stream>>>(agg2, cnt2, h2, A2);
    sage_gemm_bf16_wmma<<<dim3(MP2 / 128, NP2 / 64), dim3(256), 0, stream>>>(A2, Wc2, bp2, logits, KA12, NP2, 0);

    // ---- log_softmax -> d_out (4096 x 47) ----
    sage_logsoftmax_kernel<<<dim3(cN3), dim3(32), 0, stream>>>(logits, out);
}
